// MixtureOfExperts_57019985822184
// MI455X (gfx1250) — matrix operand order repaired and checked by reading, not compiled, not run on hardware
//
#include <hip/hip_runtime.h>
#include <hip/hip_bf16.h>
#include <math.h>

// Problem constants (B=2, S=1024, H=1024, I=2048, E=8, TOP_K=2)
#define T_TOK 2048
#define HDIM  1024
#define IDIM  2048
#define NEXP  8

typedef __attribute__((ext_vector_type(16))) __bf16 v16bf;
typedef __attribute__((ext_vector_type(8)))  float  v8f;
typedef __attribute__((ext_vector_type(8)))  __bf16 bf16x8;
typedef __attribute__((ext_vector_type(4)))  __bf16 bf16x4;

// ---------------------------------------------------------------------------
// Async global->LDS staging (CDNA5 GLOBAL_LOAD_ASYNC_TO_LDS_B64, ASYNCcnt)
// LDS dest operand is a per-lane VGPR holding the LDS byte address; the low
// 32 bits of a flat shared pointer ARE the LDS offset (aperture truncation).
// ---------------------------------------------------------------------------
__device__ __forceinline__ void async_load_b64(void* lds_ptr, const void* gptr) {
    unsigned int lo = (unsigned int)(unsigned long long)(uintptr_t)lds_ptr;
    asm volatile("global_load_async_to_lds_b64 %0, %1, off"
                 :: "v"(lo), "v"((unsigned long long)(uintptr_t)gptr)
                 : "memory");
}
__device__ __forceinline__ void wait_async0() {
    asm volatile("s_wait_asynccnt 0x0" ::: "memory");
}

// ---------------------------------------------------------------------------
// WMMA fragment loaders (layouts per CDNA5 ISA 7.12.2, 16-bit A 16x32 / B 32x16)
// ---------------------------------------------------------------------------
__device__ __forceinline__ v16bf load_frag_a(const __bf16* As, int row0, int lane, int lda) {
    int r  = row0 + (lane & 15);
    int kb = (lane < 16) ? 0 : 8;
    const __bf16* p = As + r * lda;
    bf16x8 lo = *(const bf16x8*)(p + kb);        // k = kb .. kb+7      (16B)
    bf16x8 hi = *(const bf16x8*)(p + kb + 16);   // k = kb+16 .. kb+23  (16B)
    v16bf a;
#pragma unroll
    for (int i = 0; i < 8; ++i) { a[i] = lo[i]; a[i + 8] = hi[i]; }
    return a;
}

// B tile in LDS stored TRANSPOSED: [n][k] -> k = kb..kb+7 and kb+16..kb+23 per lane (2x16B).
__device__ __forceinline__ v16bf load_frag_b(const __bf16* Bs, int col0, int lane, int ldb) {
    int n  = col0 + (lane & 15);
    int kb = (lane < 16) ? 0 : 8;
    const __bf16* p = Bs + n * ldb + kb;
    bf16x8 lo = *(const bf16x8*)(p);
    bf16x8 hi = *(const bf16x8*)(p + 16);
    v16bf b;
#pragma unroll
    for (int i = 0; i < 8; ++i) { b[i] = lo[i]; b[i + 8] = hi[i]; }
    return b;
}

// ---------------------------------------------------------------------------
// Kernel 1: gating — logits, softmax, top-2, renormalize, count per expert
// ---------------------------------------------------------------------------
__global__ void moe_gate(const float* __restrict__ x, const float* __restrict__ gw,
                         const float* __restrict__ gb, int* __restrict__ cnt,
                         int* __restrict__ top_idx, float* __restrict__ top_w) {
    int t = blockIdx.x * blockDim.x + threadIdx.x;
    if (t >= T_TOK) return;
    float lg[NEXP];
#pragma unroll
    for (int e = 0; e < NEXP; ++e) lg[e] = gb[e];
    const float* xt = x + (size_t)t * HDIM;
    for (int h = 0; h < HDIM; ++h) {
        float xv = xt[h];
        const float* g = gw + h * NEXP;   // wave-uniform row -> scalar loads
#pragma unroll
        for (int e = 0; e < NEXP; ++e) lg[e] = fmaf(xv, g[e], lg[e]);
    }
    float m = lg[0];
#pragma unroll
    for (int e = 1; e < NEXP; ++e) m = fmaxf(m, lg[e]);
    float p[NEXP];
#pragma unroll
    for (int e = 0; e < NEXP; ++e) p[e] = __expf(lg[e] - m);   // common scale cancels
    int   i1 = 0; float b1 = p[0];
#pragma unroll
    for (int e = 1; e < NEXP; ++e) if (p[e] > b1) { b1 = p[e]; i1 = e; }
    int   i2 = (i1 == 0) ? 1 : 0; float b2 = p[i2];
#pragma unroll
    for (int e = 0; e < NEXP; ++e) if (e != i1 && p[e] > b2) { b2 = p[e]; i2 = e; }
    float inv = __builtin_amdgcn_rcpf(b1 + b2);
    top_idx[t * 2 + 0] = i1; top_idx[t * 2 + 1] = i2;
    top_w  [t * 2 + 0] = b1 * inv; top_w[t * 2 + 1] = b2 * inv;
    atomicAdd(&cnt[i1], 1);
    atomicAdd(&cnt[i2], 1);
}

// Kernel 2: exclusive prefix over counts; zero phase-2 counters
__global__ void moe_offsets(const int* __restrict__ cnt, int* __restrict__ off,
                            int* __restrict__ cnt2) {
    if (threadIdx.x == 0) {
        int s = 0;
#pragma unroll
        for (int e = 0; e < NEXP; ++e) { off[e] = s; s += cnt[e]; }
    }
    if (threadIdx.x < NEXP) cnt2[threadIdx.x] = 0;
}

// Kernel 3: scatter tokens into compacted per-expert slot lists
__global__ void moe_scatter(const int* __restrict__ top_idx, const float* __restrict__ top_w,
                            const int* __restrict__ off, int* __restrict__ cnt2,
                            int* __restrict__ tok, float* __restrict__ scl) {
    int t = blockIdx.x * blockDim.x + threadIdx.x;
    if (t >= T_TOK) return;
#pragma unroll
    for (int k = 0; k < 2; ++k) {
        int e = top_idx[t * 2 + k];
        int p = atomicAdd(&cnt2[e], 1);
        int s = off[e] + p;
        tok[s] = t;
        scl[s] = top_w[t * 2 + k];
    }
}

// ---------------------------------------------------------------------------
// Kernel 4: GEMM1  h[slot, I] = silu(x·w1[e]) * (x·w3[e])   (bf16 WMMA, f32 acc)
// block = 256 threads = 8 waves; block tile 32 tokens x 128 I-cols;
// waves 2(m) x 4(n), each wave owns 16x32 -> 4 WMMAs per K-step.
// ---------------------------------------------------------------------------
__global__ void __launch_bounds__(256)
moe_gemm1(const float* __restrict__ x, const float* __restrict__ w1,
          const float* __restrict__ w3, const int* __restrict__ cnt,
          const int* __restrict__ off, const int* __restrict__ tok,
          __bf16* __restrict__ hbuf) {
    const int e = blockIdx.z;
    const int c = cnt[e];
    const int tileM = blockIdx.y;
    if (tileM * 32 >= c) return;
    const int o  = off[e];
    const int n0 = blockIdx.x * 128;

    __shared__ alignas(16) __bf16 As [32][40];
    __shared__ alignas(16) __bf16 B1s[128][40];
    __shared__ alignas(16) __bf16 B3s[128][40];

    const int lane = threadIdx.x & 31;
    const int wv   = threadIdx.x >> 5;
    const int wm   = wv & 1;        // token sub-tile (0/1)
    const int wn   = wv >> 1;       // I sub-tile (0..3), 32 cols each

    v8f acc1a = {}, acc1b = {}, acc3a = {}, acc3b = {};
    const float* w1e = w1 + (size_t)e * HDIM * IDIM;
    const float* w3e = w3 + (size_t)e * HDIM * IDIM;

    for (int k0 = 0; k0 < HDIM; k0 += 32) {
        __syncthreads();
        {   // stage A: gathered token rows, fp32 -> bf16 (v_cvt_pk_bf16_f32)
            int row = threadIdx.x >> 3;
            int kq  = (threadIdx.x & 7) << 2;
            int pos = tileM * 32 + row;
            int t   = tok[o + (pos < c ? pos : 0)];
            float4 xv = *(const float4*)(x + (size_t)t * HDIM + k0 + kq);
            bf16x4 v; v[0] = (__bf16)xv.x; v[1] = (__bf16)xv.y;
                      v[2] = (__bf16)xv.z; v[3] = (__bf16)xv.w;
            *(bf16x4*)&As[row][kq] = v;
        }
        if (k0 + 32 < HDIM) {   // prefetch next weight tiles (global_prefetch_b8)
            __builtin_prefetch((const void*)(w1e + (size_t)(k0 + 32) * IDIM + n0), 0, 1);
            __builtin_prefetch((const void*)(w3e + (size_t)(k0 + 32) * IDIM + n0), 0, 1);
        }
#pragma unroll
        for (int j = 0; j < 4; ++j) {  // stage B1/B3: 32k x 128n, transpose into [n][k]
            int q  = threadIdx.x + j * 256;      // 0..1023
            int kk = q >> 5;                     // 0..31
            int nq = (q & 31) << 2;              // 0..124
            float4 a4 = *(const float4*)(w1e + (size_t)(k0 + kk) * IDIM + n0 + nq);
            float4 b4 = *(const float4*)(w3e + (size_t)(k0 + kk) * IDIM + n0 + nq);
            B1s[nq + 0][kk] = (__bf16)a4.x; B1s[nq + 1][kk] = (__bf16)a4.y;
            B1s[nq + 2][kk] = (__bf16)a4.z; B1s[nq + 3][kk] = (__bf16)a4.w;
            B3s[nq + 0][kk] = (__bf16)b4.x; B3s[nq + 1][kk] = (__bf16)b4.y;
            B3s[nq + 2][kk] = (__bf16)b4.z; B3s[nq + 3][kk] = (__bf16)b4.w;
        }
        __syncthreads();
        v16bf af   = load_frag_a(&As [0][0], wm * 16, lane, 40);
        v16bf b1f0 = load_frag_b(&B1s[0][0], wn * 32,      lane, 40);
        v16bf b1f1 = load_frag_b(&B1s[0][0], wn * 32 + 16, lane, 40);
        v16bf b3f0 = load_frag_b(&B3s[0][0], wn * 32,      lane, 40);
        v16bf b3f1 = load_frag_b(&B3s[0][0], wn * 32 + 16, lane, 40);
        acc1a = __builtin_amdgcn_wmma_f32_16x16x32_bf16(false, af, false, b1f0,
                                                        (short)0, acc1a, false, false);
        acc1b = __builtin_amdgcn_wmma_f32_16x16x32_bf16(false, af, false, b1f1,
                                                        (short)0, acc1b, false, false);
        acc3a = __builtin_amdgcn_wmma_f32_16x16x32_bf16(false, af, false, b3f0,
                                                        (short)0, acc3a, false, false);
        acc3b = __builtin_amdgcn_wmma_f32_16x16x32_bf16(false, af, false, b3f1,
                                                        (short)0, acc3b, false, false);
    }
    // epilogue: h = silu(g) * u with fast rcp, store bf16
    int mloc = (lane < 16) ? 0 : 8;
    int nloc = lane & 15;
#pragma unroll
    for (int sub = 0; sub < 2; ++sub) {
        int icol = n0 + wn * 32 + sub * 16 + nloc;
#pragma unroll
        for (int r = 0; r < 8; ++r) {
            int pos = tileM * 32 + wm * 16 + r + mloc;
            if (pos < c) {
                float g = sub ? acc1b[r] : acc1a[r];
                float u = sub ? acc3b[r] : acc3a[r];
                float h = g * __builtin_amdgcn_rcpf(1.0f + __expf(-g)) * u;
                hbuf[(size_t)(o + pos) * IDIM + icol] = (__bf16)h;
            }
        }
    }
}

// ---------------------------------------------------------------------------
// Kernel 5: GEMM2  out[t, :] += scale * h[slot, :] · w2[e]
// A tile (already bf16) staged via GLOBAL_LOAD_ASYNC_TO_LDS_B64 (ASYNCcnt).
// block tile 32 slots x 128 H cols; each wave 16x32 -> 2 WMMAs per K-step.
// ---------------------------------------------------------------------------
__global__ void __launch_bounds__(256)
moe_gemm2(const __bf16* __restrict__ hbuf, const float* __restrict__ w2,
          const int* __restrict__ cnt, const int* __restrict__ off,
          const int* __restrict__ tok, const float* __restrict__ scl,
          float* __restrict__ out) {
    const int e = blockIdx.z;
    const int c = cnt[e];
    const int tileM = blockIdx.y;
    if (tileM * 32 >= c) return;
    const int o  = off[e];
    const int n0 = blockIdx.x * 128;

    __shared__ alignas(16) __bf16 As[32][40];
    __shared__ alignas(16) __bf16 Bs[128][40];

    const int lane = threadIdx.x & 31;
    const int wv   = threadIdx.x >> 5;
    const int wm   = wv & 1;
    const int wn   = wv >> 1;

    v8f acca = {}, accb = {};
    const float* w2e = w2 + (size_t)e * IDIM * HDIM;

    // per-thread A-staging coordinates (32 rows x 32 k bf16 = 256 x 8B)
    const int arow = threadIdx.x >> 3;
    const int ako  = (threadIdx.x & 7) << 2;          // 4 bf16 = 8 bytes
    const int apos = tileM * 32 + arow;
    const int asg  = o + (apos < c ? apos : 0);
    const __bf16* arow_ptr = hbuf + (size_t)asg * IDIM + ako;

    for (int k0 = 0; k0 < IDIM; k0 += 32) {
        __syncthreads();
        // stage A: async DMA global(bf16) -> LDS, no VGPR round-trip
        async_load_b64(&As[arow][ako], arow_ptr + k0);
        if (k0 + 32 < IDIM)
            __builtin_prefetch((const void*)(w2e + (size_t)(k0 + 32) * HDIM + n0), 0, 1);
#pragma unroll
        for (int j = 0; j < 4; ++j) {  // stage B: w2 fp32 -> bf16, transposed [n][k]
            int q  = threadIdx.x + j * 256;
            int kk = q >> 5;
            int nq = (q & 31) << 2;
            float4 a4 = *(const float4*)(w2e + (size_t)(k0 + kk) * HDIM + n0 + nq);
            Bs[nq + 0][kk] = (__bf16)a4.x; Bs[nq + 1][kk] = (__bf16)a4.y;
            Bs[nq + 2][kk] = (__bf16)a4.z; Bs[nq + 3][kk] = (__bf16)a4.w;
        }
        wait_async0();              // ASYNCcnt==0: A tile resident in LDS
        __syncthreads();
        v16bf af  = load_frag_a(&As[0][0], wm * 16, lane, 40);
        v16bf bf0 = load_frag_b(&Bs[0][0], wn * 32,      lane, 40);
        v16bf bf1 = load_frag_b(&Bs[0][0], wn * 32 + 16, lane, 40);
        acca = __builtin_amdgcn_wmma_f32_16x16x32_bf16(false, af, false, bf0,
                                                       (short)0, acca, false, false);
        accb = __builtin_amdgcn_wmma_f32_16x16x32_bf16(false, af, false, bf1,
                                                       (short)0, accb, false, false);
    }
    int mloc = (lane < 16) ? 0 : 8;
    int nloc = lane & 15;
#pragma unroll
    for (int sub = 0; sub < 2; ++sub) {
        int hcol = n0 + wn * 32 + sub * 16 + nloc;
#pragma unroll
        for (int r = 0; r < 8; ++r) {
            int pos = tileM * 32 + wm * 16 + r + mloc;
            if (pos < c) {
                int   sg = o + pos;
                int   t  = tok[sg];
                float s  = scl[sg];
                float v  = sub ? accb[r] : acca[r];
                atomicAdd(out + (size_t)t * HDIM + hcol, s * v);
            }
        }
    }
}

// ---------------------------------------------------------------------------
// Host launcher
// ---------------------------------------------------------------------------
extern "C" void kernel_launch(void* const* d_in, const int* in_sizes, int n_in,
                              void* d_out, int out_size, void* d_ws, size_t ws_size,
                              hipStream_t stream) {
    const float* x  = (const float*)d_in[0];
    const float* gw = (const float*)d_in[1];
    const float* gb = (const float*)d_in[2];
    const float* w1 = (const float*)d_in[3];
    const float* w3 = (const float*)d_in[4];
    const float* w2 = (const float*)d_in[5];
    float* out = (float*)d_out;

    char* ws = (char*)d_ws;
    int*    cnt  = (int*)  (ws + 0);            // 32 B
    int*    cnt2 = (int*)  (ws + 32);           // 32 B
    int*    off  = (int*)  (ws + 64);           // 32 B
    int*    tidx = (int*)  (ws + 128);          // T*2*4 = 16384 B
    float*  tw   = (float*)(ws + 16512);        // 16384 B
    int*    tok  = (int*)  (ws + 32896);        // 16384 B
    float*  scl  = (float*)(ws + 49280);        // 16384 B
    __bf16* hbuf = (__bf16*)(ws + 65792);       // 2T * I * 2 = 16 MiB

    hipMemsetAsync(ws, 0, 64, stream);                              // cnt, cnt2
    hipMemsetAsync(d_out, 0, sizeof(float) * T_TOK * HDIM, stream); // accumulate target

    moe_gate   <<<dim3((T_TOK + 255) / 256), dim3(256), 0, stream>>>(x, gw, gb, cnt, tidx, tw);
    moe_offsets<<<dim3(1),                   dim3(32),  0, stream>>>(cnt, off, cnt2);
    moe_scatter<<<dim3((T_TOK + 255) / 256), dim3(256), 0, stream>>>(tidx, tw, off, cnt2, tok, scl);
    moe_gemm1  <<<dim3(IDIM / 128, T_TOK / 32, NEXP), dim3(256), 0, stream>>>(x, w1, w3, cnt, off, tok, hbuf);
    moe_gemm2  <<<dim3(HDIM / 128, T_TOK / 32, NEXP), dim3(256), 0, stream>>>(hbuf, w2, cnt, off, tok, scl, out);
}
